// DisPU_86251533238922
// MI455X (gfx1250) — matrix operand order repaired and checked by reading, not compiled, not run on hardware
//
#include <hip/hip_runtime.h>
#include <hip/hip_bf16.h>
#include <math.h>

// =====================================================================
// DisPU forward for MI455X (gfx1250, wave32, WMMA bf16).
// All matrix math lowered to v_wmma_f32_16x16x32_bf16 (f32 accumulate).
// Attention is reassociated (no softmax): val = (V*K^T)*Q, avoiding the
// [B,M,M] matrix entirely.
// GEMM: 2x2 register tiling (32x32 output per wave), clamped-address
// unconditional loads (OOB rows/cols are masked at store; only the K
// tail needs value zeroing), peeled K-tail, branch-free inner loop with
// two-tile-ahead global_prefetch_b8.
// =====================================================================

typedef __attribute__((ext_vector_type(16))) __bf16 v16bf;
typedef __attribute__((ext_vector_type(8)))  float  v8f;

static const int BT = 2;      // batch
static const int NN = 1024;   // input points
static const int KK = 16;     // neighbors
static const int MM = 4096;   // upsampled points (UP=4)
static const int GS = 64;     // sqrt(MM) grid code

union BfVec { v16bf v; __bf16 e[16]; };

// ---------------------------------------------------------------------
// Generic strided-batched GEMM: C[o,n] = sum_c A(o,c)*B(c,n) (+bias)(+relu)
// A(o,c) at A[o*saR + c*saC + bz*saB]   (usually weights, saB=0)
// B(c,n) at B[c*sbR + n*sbC + bz*sbB]   (activations)
// C(o,n) at C[o*scR + n*scC + bz*scB]
// One wave -> 32x32 output tile (4 accumulators); K stepped by 32.
// VGPR layouts follow CDNA5 ISA 7.12.2 (wave32).
// ---------------------------------------------------------------------
__global__ void __launch_bounds__(128)
gemm_wmma_kernel(const float* __restrict__ A, long saR, long saC, long saB,
                 const float* __restrict__ Bm, long sbR, long sbC, long sbB,
                 float* __restrict__ Cm, long scR, long scC, long scB,
                 const float* __restrict__ bias,
                 int O, int Ncols, int Cd, int relu)
{
  const int lane = threadIdx.x;              // 0..31
  const int half = lane >> 4, l16 = lane & 15;
  const int n0 = (blockIdx.x * blockDim.y + threadIdx.y) * 32;
  const int m0 = blockIdx.y * 32;
  if (n0 >= Ncols) return;                   // wave-uniform (EXEC stays all-1)
  const long bz = blockIdx.z;
  const float* Ab = A  + bz * saB;
  const float* Bb = Bm + bz * sbB;
  float*       Cb = Cm + bz * scB;

  // Clamp row/col indices for memory safety only: an OOB A-row corrupts
  // only D-rows >= O, an OOB B-col corrupts only D-cols >= Ncols, and
  // those are never stored. No value masking needed in the main loop.
  int am0 = m0 + l16;       if (am0 > O - 1)     am0 = O - 1;
  int am1 = m0 + 16 + l16;  if (am1 > O - 1)     am1 = O - 1;
  int cn0 = n0 + l16;       if (cn0 > Ncols - 1) cn0 = Ncols - 1;
  int cn1 = n0 + 16 + l16;  if (cn1 > Ncols - 1) cn1 = Ncols - 1;
  const float* aR0 = Ab + (long)am0 * saR;
  const float* aR1 = Ab + (long)am1 * saR;
  const float* bC0 = Bb + (long)cn0 * sbC;
  const float* bC1 = Bb + (long)cn1 * sbC;

  const v8f vz = {0.f,0.f,0.f,0.f,0.f,0.f,0.f,0.f};
  v8f acc00 = vz, acc01 = vz, acc10 = vz, acc11 = vz;

  const int nFull = Cd & ~31;
  int c0 = 0;
  for (; c0 < nFull; c0 += 32) {             // branch-free steady state
    // Prefetch two K-tiles ahead (clamped -> unconditional, no branch).
    // 32 lanes hit 32 distinct rows/cols; each pulls a full cacheline.
    {
      long cp = (c0 + 64 < Cd) ? (long)(c0 + 64) : 0L;
      __builtin_prefetch(&aR0[cp * saC], 0, 1);
      __builtin_prefetch(&aR1[cp * saC], 0, 1);
      __builtin_prefetch(&bC0[cp * sbR], 0, 1);
      __builtin_prefetch(&bC1[cp * sbR], 0, 1);
    }
    BfVec a0, a1, b0, b1;
#pragma unroll
    for (int i = 0; i < 8; i++) {
      // A 16x32 bf16: VGPR i<4 -> K=2i+8*half ; i>=4 -> K=16+2(i-4)+8*half
      int kb = (i < 4) ? (2*i + half*8) : (16 + 2*(i-4) + half*8);
      long o0 = (long)(c0 + kb) * saC;
      long o1 = (long)(c0 + kb + 1) * saC;
      a0.e[2*i]   = (__bf16)aR0[o0];
      a0.e[2*i+1] = (__bf16)aR0[o1];
      a1.e[2*i]   = (__bf16)aR1[o0];
      a1.e[2*i+1] = (__bf16)aR1[o1];
    }
#pragma unroll
    for (int j = 0; j < 8; j++) {
#pragma unroll
      for (int h = 0; h < 2; h++) {
        // B 32x16 bf16: lane=N col; K = 2j+h+8*half (+8 more for j>=4)
        long o = (long)(c0 + 2*j + h + half*8 + (j >= 4 ? 8 : 0)) * sbR;
        b0.e[2*j+h] = (__bf16)bC0[o];
        b1.e[2*j+h] = (__bf16)bC1[o];
      }
    }
    acc00 = __builtin_amdgcn_wmma_f32_16x16x32_bf16(false, a0.v, false, b0.v,
                                                    (short)0, acc00, false, false);
    acc01 = __builtin_amdgcn_wmma_f32_16x16x32_bf16(false, a0.v, false, b1.v,
                                                    (short)0, acc01, false, false);
    acc10 = __builtin_amdgcn_wmma_f32_16x16x32_bf16(false, a1.v, false, b0.v,
                                                    (short)0, acc10, false, false);
    acc11 = __builtin_amdgcn_wmma_f32_16x16x32_bf16(false, a1.v, false, b1.v,
                                                    (short)0, acc11, false, false);
  }
  if (c0 < Cd) {                             // peeled K tail: zero k>=Cd
    BfVec a0, a1, b0, b1;
#pragma unroll
    for (int i = 0; i < 8; i++) {
      int kb = (i < 4) ? (2*i + half*8) : (16 + 2*(i-4) + half*8);
      int k0 = c0 + kb, k1 = c0 + kb + 1;
      int k0c = k0 < Cd ? k0 : Cd - 1;
      int k1c = k1 < Cd ? k1 : Cd - 1;
      float f00 = aR0[(long)k0c * saC]; if (k0 >= Cd) f00 = 0.f;
      float f01 = aR0[(long)k1c * saC]; if (k1 >= Cd) f01 = 0.f;
      float f10 = aR1[(long)k0c * saC]; if (k0 >= Cd) f10 = 0.f;
      float f11 = aR1[(long)k1c * saC]; if (k1 >= Cd) f11 = 0.f;
      a0.e[2*i] = (__bf16)f00; a0.e[2*i+1] = (__bf16)f01;
      a1.e[2*i] = (__bf16)f10; a1.e[2*i+1] = (__bf16)f11;
    }
#pragma unroll
    for (int j = 0; j < 8; j++) {
#pragma unroll
      for (int h = 0; h < 2; h++) {
        int k = c0 + 2*j + h + half*8 + (j >= 4 ? 8 : 0);
        int kc = k < Cd ? k : Cd - 1;
        float g0 = bC0[(long)kc * sbR]; if (k >= Cd) g0 = 0.f;
        float g1 = bC1[(long)kc * sbR]; if (k >= Cd) g1 = 0.f;
        b0.e[2*j+h] = (__bf16)g0;
        b1.e[2*j+h] = (__bf16)g1;
      }
    }
    acc00 = __builtin_amdgcn_wmma_f32_16x16x32_bf16(false, a0.v, false, b0.v,
                                                    (short)0, acc00, false, false);
    acc01 = __builtin_amdgcn_wmma_f32_16x16x32_bf16(false, a0.v, false, b1.v,
                                                    (short)0, acc01, false, false);
    acc10 = __builtin_amdgcn_wmma_f32_16x16x32_bf16(false, a1.v, false, b0.v,
                                                    (short)0, acc10, false, false);
    acc11 = __builtin_amdgcn_wmma_f32_16x16x32_bf16(false, a1.v, false, b1.v,
                                                    (short)0, acc11, false, false);
  }

  // C/D layout: VGPR r -> row (r + 8*half), col l16
#pragma unroll
  for (int r = 0; r < 8; r++) {
    int mA = m0 + r + half*8, mB = mA + 16;
    int nA = n0 + l16,        nB = nA + 16;
    float v;
    if (mA < O && nA < Ncols) {
      v = acc00[r]; if (bias) v += bias[mA]; if (relu) v = v > 0.f ? v : 0.f;
      Cb[(long)mA*scR + (long)nA*scC] = v;
    }
    if (mA < O && nB < Ncols) {
      v = acc01[r]; if (bias) v += bias[mA]; if (relu) v = v > 0.f ? v : 0.f;
      Cb[(long)mA*scR + (long)nB*scC] = v;
    }
    if (mB < O && nA < Ncols) {
      v = acc10[r]; if (bias) v += bias[mB]; if (relu) v = v > 0.f ? v : 0.f;
      Cb[(long)mB*scR + (long)nA*scC] = v;
    }
    if (mB < O && nB < Ncols) {
      v = acc11[r]; if (bias) v += bias[mB]; if (relu) v = v > 0.f ? v : 0.f;
      Cb[(long)mB*scR + (long)nB*scC] = v;
    }
  }
}

// ---------------------------------------------------------------------
// Training-mode BatchNorm: stats over (batch, spatial) per channel.
// x layout: [B][C][L], element (b,c,l) at b*bStride + c*L + l.
// ---------------------------------------------------------------------
__global__ void bn_stats_kernel(const float* __restrict__ x, float* mean, float* var,
                                int C, long L, long bStride, int Bn)
{
  __shared__ float ssum[256], ssq[256];
  const int c = blockIdx.x;
  float s = 0.f, q = 0.f;
  const long tot = (long)Bn * L;
  for (long i = threadIdx.x; i < tot; i += blockDim.x) {
    long b = i / L, l = i - b * L;
    float v = x[b*bStride + (long)c*L + l];
    s += v; q += v*v;
  }
  ssum[threadIdx.x] = s; ssq[threadIdx.x] = q; __syncthreads();
  for (int st = 128; st > 0; st >>= 1) {
    if ((int)threadIdx.x < st) { ssum[threadIdx.x] += ssum[threadIdx.x+st];
                                 ssq[threadIdx.x]  += ssq[threadIdx.x+st]; }
    __syncthreads();
  }
  if (threadIdx.x == 0) {
    float mu = ssum[0] / (float)tot;
    mean[c] = mu;
    var[c]  = ssq[0] / (float)tot - mu*mu;
  }
}

__global__ void bn_apply_kernel(float* __restrict__ x, const float* mean, const float* var,
                                const float* g, const float* be,
                                int C, long L, long bStride, long total, int relu)
{
  for (long i = blockIdx.x*(long)blockDim.x + threadIdx.x; i < total;
       i += (long)gridDim.x*blockDim.x) {
    long b = i / ((long)C*L);
    long rem = i - b*(long)C*L;
    int  c = (int)(rem / L);
    long l = rem - (long)c*L;
    long off = b*bStride + (long)c*L + l;
    float v = (x[off] - mean[c]) * rsqrtf(var[c] + 1e-5f) * g[c] + be[c];
    if (relu) v = v > 0.f ? v : 0.f;
    x[off] = v;
  }
}

// ---------------------------------------------------------------------
// Brute-force KNN (k=16 + self). One thread per query point; top-17
// insertion sort; self (d2==0) lands first and is dropped.
// ---------------------------------------------------------------------
__global__ void knn_kernel(const float* __restrict__ pts, int* __restrict__ idx,
                           int Bn, int Np)
{
  long t = blockIdx.x*(long)blockDim.x + threadIdx.x;
  if (t >= (long)Bn*Np) return;
  int b = (int)(t / Np), m = (int)(t - (long)b*Np);
  const float* P = pts + (long)b*Np*3;
  float qx = P[m*3+0], qy = P[m*3+1], qz = P[m*3+2];
  float best[17]; int bi[17];
  for (int i = 0; i < 17; i++) { best[i] = 3.0e38f; bi[i] = 0; }
  for (int n = 0; n < Np; n++) {
    float dx = P[n*3+0]-qx, dy = P[n*3+1]-qy, dz = P[n*3+2]-qz;
    float d2 = dx*dx + dy*dy + dz*dz;
    if (d2 < best[16]) {
      int j = 16;
      while (j > 0 && best[j-1] > d2) { best[j] = best[j-1]; bi[j] = bi[j-1]; j--; }
      best[j] = d2; bi[j] = n;
    }
  }
  for (int k = 0; k < 16; k++) idx[t*16 + k] = bi[k+1];
}

// ---------------------------------------------------------------------
// Elementwise / data-movement kernels (grid-stride).
// ---------------------------------------------------------------------
__global__ void gather_kernel(const float* __restrict__ src, const int* __restrict__ idx,
                              float* __restrict__ out, int C, int Np, int Kn,
                              long srcB, long outB, long total)
{ // out[b][c][n*Kn+k] = src[b][c][idx[b][n][k]]
  for (long i = blockIdx.x*(long)blockDim.x + threadIdx.x; i < total;
       i += (long)gridDim.x*blockDim.x) {
    long b = i / ((long)C*Np*Kn);
    long rem = i - b*(long)C*Np*Kn;
    int  c = (int)(rem / ((long)Np*Kn));
    long r2 = rem - (long)c*Np*Kn;
    int  n = (int)(r2 / Kn), k = (int)(r2 - (long)n*Kn);
    int  j = idx[(b*Np + n)*Kn + k];
    out[b*outB + (long)c*Np*Kn + r2] = src[b*srcB + (long)c*Np + j];
  }
}

__global__ void broadcast_kernel(const float* __restrict__ src, float* __restrict__ out,
                                 int C, int Np, int Kn, long srcB, long outB, long total)
{ // out[b][c][n*Kn+k] = src[b][c][n]
  for (long i = blockIdx.x*(long)blockDim.x + threadIdx.x; i < total;
       i += (long)gridDim.x*blockDim.x) {
    long b = i / ((long)C*Np*Kn);
    long rem = i - b*(long)C*Np*Kn;
    int  c = (int)(rem / ((long)Np*Kn));
    long r2 = rem - (long)c*Np*Kn;
    int  n = (int)(r2 / Kn);
    out[b*outB + (long)c*Np*Kn + r2] = src[b*srcB + (long)c*Np + n];
  }
}

__global__ void copy_ch_kernel(const float* __restrict__ src, float* __restrict__ out,
                               int C, long L, long srcB, long outB, long total)
{ // out[b][c][l] = src[b][c][l]
  for (long i = blockIdx.x*(long)blockDim.x + threadIdx.x; i < total;
       i += (long)gridDim.x*blockDim.x) {
    long b = i / ((long)C*L);
    long rem = i - b*(long)C*L;
    out[b*outB + rem] = src[b*srcB + rem];
  }
}

__global__ void maxpool_kernel(const float* __restrict__ in, float* __restrict__ out,
                               int C, int Np, int Kn, long inB, long outB, long total)
{ // out[b][c][n] = max_k in[b][c][n*Kn+k]
  for (long i = blockIdx.x*(long)blockDim.x + threadIdx.x; i < total;
       i += (long)gridDim.x*blockDim.x) {
    long b = i / ((long)C*Np);
    long rem = i - b*(long)C*Np;
    int  c = (int)(rem / Np), n = (int)(rem - (long)c*Np);
    const float* p = in + b*inB + ((long)c*Np + n)*Kn;
    float mx = p[0];
    for (int k = 1; k < Kn; k++) mx = p[k] > mx ? p[k] : mx;
    out[b*outB + rem] = mx;
  }
}

__global__ void pts_to_chans_kernel(const float* __restrict__ src, float* __restrict__ dst,
                                    int C, int Np, long total)
{ // src [B,Np,C] -> dst [B,C,Np]
  for (long i = blockIdx.x*(long)blockDim.x + threadIdx.x; i < total;
       i += (long)gridDim.x*blockDim.x) {
    long b = i / ((long)C*Np);
    long rem = i - b*(long)C*Np;
    int  c = (int)(rem / Np), n = (int)(rem - (long)c*Np);
    dst[i] = src[(b*Np + n)*(long)C + c];
  }
}

__global__ void chans_to_pts_kernel(const float* __restrict__ src, float* __restrict__ dst,
                                    float* __restrict__ dst2, int C, int Np, int mode,
                                    long total)
{ // src [B,C,Np] -> dst [B,Np,C] ; mode 1: sigmoid(x)-0.5
  for (long i = blockIdx.x*(long)blockDim.x + threadIdx.x; i < total;
       i += (long)gridDim.x*blockDim.x) {
    long b = i / ((long)Np*C);
    long rem = i - b*(long)Np*C;
    int  n = (int)(rem / C), c = (int)(rem - (long)n*C);
    float v = src[(b*C + c)*(long)Np + n];
    if (mode == 1) v = 1.f/(1.f + expf(-v)) - 0.5f;
    dst[i] = v;
    if (dst2) dst2[i] = v;
  }
}

__global__ void expand_feat_kernel(const float* __restrict__ y, float* __restrict__ feat,
                                   int Cy, int Np, int Mp, long total)
{ // feat[b][c][m]: c<Cy -> tile of y; last 2 channels -> 2D grid code
  for (long i = blockIdx.x*(long)blockDim.x + threadIdx.x; i < total;
       i += (long)gridDim.x*blockDim.x) {
    long b = i / ((long)(Cy+2)*Mp);
    long rem = i - b*(long)(Cy+2)*Mp;
    int  c = (int)(rem / Mp), m = (int)(rem - (long)c*Mp);
    float v;
    if (c < Cy) v = y[(b*(long)Cy + c)*Np + (m % Np)];
    else {
      int ii = m / GS, jj = m - (m / GS)*GS;
      int t = (c == Cy) ? ii : jj;
      v = -0.2f + 0.4f * (float)t / (float)(GS - 1);
    }
    feat[i] = v;
  }
}

__global__ void gdiff_kernel(const float* __restrict__ coordP, const int* __restrict__ idx2,
                             float* __restrict__ out, int Mp, int Kn, long total)
{ // out[b][d][m*Kn+k] = coordP[b][idx][d] - coordP[b][m][d]
  for (long i = blockIdx.x*(long)blockDim.x + threadIdx.x; i < total;
       i += (long)gridDim.x*blockDim.x) {
    long b = i / (3L*Mp*Kn);
    long rem = i - b*3L*Mp*Kn;
    int  d = (int)(rem / ((long)Mp*Kn));
    long r2 = rem - (long)d*Mp*Kn;
    int  m = (int)(r2 / Kn), k = (int)(r2 - (long)m*Kn);
    int  pidx = idx2[(b*Mp + m)*Kn + k];
    out[i] = coordP[(b*Mp + pidx)*3 + d] - coordP[(b*Mp + m)*3 + d];
  }
}

__global__ void wsum_kernel(const float* __restrict__ w16, float* __restrict__ wsum,
                            int Cw, int Mp, int Kn, long total)
{ // wsum[b][j][m] = sum_k w16[b][j][m*Kn+k]
  for (long i = blockIdx.x*(long)blockDim.x + threadIdx.x; i < total;
       i += (long)gridDim.x*blockDim.x) {
    long b = i / ((long)Cw*Mp);
    long rem = i - b*(long)Cw*Mp;
    int  j = (int)(rem / Mp), m = (int)(rem - (long)j*Mp);
    const float* p = w16 + ((b*(long)Cw + j)*Mp + m)*Kn;
    float s = 0.f;
    for (int k = 0; k < Kn; k++) s += p[k];
    wsum[i] = s;
  }
}

__global__ void rw_add_kernel(const float* __restrict__ wsum, const float* __restrict__ f,
                              const float* __restrict__ mp, float* __restrict__ out,
                              int C, int Mp, int Kn, long fB, long total)
{ // out[b][c][m] = mp[b][c][m] + sum_j wsum[b][j][m] * f[b*fB + (c*Mp+m)*Kn + j]
  for (long i = blockIdx.x*(long)blockDim.x + threadIdx.x; i < total;
       i += (long)gridDim.x*blockDim.x) {
    long b = i / ((long)C*Mp);
    long rem = i - b*(long)C*Mp;
    int  c = (int)(rem / Mp), m = (int)(rem - (long)c*Mp);
    const float* wb = wsum + b*(long)Kn*Mp;
    const float* fb = f + b*fB + ((long)c*Mp + m)*Kn;
    float s = mp[i];
    for (int j = 0; j < Kn; j++) s += wb[(long)j*Mp + m] * fb[j];
    out[i] = s;
  }
}

__global__ void add_kernel(const float* __restrict__ a, const float* __restrict__ b,
                           float* __restrict__ out, long total)
{
  for (long i = blockIdx.x*(long)blockDim.x + threadIdx.x; i < total;
       i += (long)gridDim.x*blockDim.x)
    out[i] = a[i] + b[i];
}

// =====================================================================
// Host orchestration
// =====================================================================
static inline dim3 gsz(long total) {
  long b = (total + 255) >> 8;
  if (b > 32768) b = 32768;
  return dim3((unsigned)b);
}

static void gemm(hipStream_t st, const float* A, const float* Bm, float* C,
                 int O, int Nc, int Cd, int Bz,
                 long saR, long saC, long saB,
                 long sbR, long sbC, long sbB,
                 long scR, long scC, long scB,
                 const float* bias, int relu)
{
  dim3 blk(32, 4, 1);
  dim3 grd((unsigned)((Nc + 127) / 128), (unsigned)((O + 31) / 32), (unsigned)Bz);
  gemm_wmma_kernel<<<grd, blk, 0, st>>>(A, saR, saC, saB, Bm, sbR, sbC, sbB,
                                        C, scR, scC, scB, bias, O, Nc, Cd, relu);
}

// Standard layout: W [O,Cd] row-major shared; X [B][Cd][Nc]; Y [B][O][Nc]
static void gemm_std(hipStream_t st, const float* W, const float* X, float* Y,
                     int O, int Nc, int Cd, int Bz, long xB, long yB,
                     const float* bias, int relu)
{
  gemm(st, W, X, Y, O, Nc, Cd, Bz, Cd, 1, 0, Nc, 1, xB, Nc, 1, yB, bias, relu);
}

static void bn_relu(hipStream_t st, float* xp, int C, long L, long bStride,
                    const float* g, const float* be, float* mean, float* var, int relu)
{
  bn_stats_kernel<<<dim3((unsigned)C), dim3(256), 0, st>>>(xp, mean, var, C, L, bStride, BT);
  long total = (long)BT * C * L;
  bn_apply_kernel<<<gsz(total), 256, 0, st>>>(xp, mean, var, g, be, C, L, bStride, total, relu);
}

// FEC weight layout within d_in (nested dict keys sorted):
// 0 m1_b 1 m1_g 2 m1_w 3 m2_b 4 m2_g 5 m2_w 6 m3_w 7 pre_b 8 pre_be 9 pre_g 10 pre_w
static void fec_step(hipStream_t st, void* const* d_in, int base, int Cin,
                     const float* xin, float* xout, float* pre24, float* y84,
                     float* mean, float* var, const int* idx1)
{
  const float* m1_b  = (const float*)d_in[base+0];
  const float* m1_g  = (const float*)d_in[base+1];
  const float* m1_w  = (const float*)d_in[base+2];
  const float* m2_b  = (const float*)d_in[base+3];
  const float* m2_g  = (const float*)d_in[base+4];
  const float* m2_w  = (const float*)d_in[base+5];
  const float* m3_w  = (const float*)d_in[base+6];
  const float* pre_b = (const float*)d_in[base+7];
  const float* pre_be= (const float*)d_in[base+8];
  const float* pre_g = (const float*)d_in[base+9];
  const float* pre_w = (const float*)d_in[base+10];
  const int Cout = 84 + Cin;
  const long NK = (long)NN * KK;

  // pre = relu(bn(conv1(x) + b))
  gemm_std(st, pre_w, xin, pre24, 24, NN, Cin, BT, (long)Cin*NN, 24L*NN, pre_b, 0);
  bn_relu(st, pre24, 24, NN, 24L*NN, pre_g, pre_be, mean, var, 1);

  // y84 channel map: [0-11 m3][12-23 m2][24-35 m1][36-59 pre bcast][60-83 gathered]
  long t = (long)BT*24*NN*KK;
  broadcast_kernel<<<gsz(t), 256, 0, st>>>(pre24, y84 + 36*NK, 24, NN, KK, 24L*NN, 84L*NK, t);
  gather_kernel  <<<gsz(t), 256, 0, st>>>(pre24, idx1, y84 + 60*NK, 24, NN, KK, 24L*NN, 84L*NK, t);

  gemm_std(st, m1_w, y84 + 36*NK, y84 + 24*NK, 12, (int)NK, 48, BT, 84L*NK, 84L*NK, nullptr, 0);
  bn_relu(st, y84 + 24*NK, 12, NK, 84L*NK, m1_g, m1_b, mean, var, 1);
  gemm_std(st, m2_w, y84 + 24*NK, y84 + 12*NK, 12, (int)NK, 60, BT, 84L*NK, 84L*NK, nullptr, 0);
  bn_relu(st, y84 + 12*NK, 12, NK, 84L*NK, m2_g, m2_b, mean, var, 1);
  gemm_std(st, m3_w, y84 + 12*NK, y84, 12, (int)NK, 72, BT, 84L*NK, 84L*NK, nullptr, 0);

  // maxpool over K -> xout[0..83]; then concat old x -> xout[84..]
  long t2 = (long)BT*84*NN;
  maxpool_kernel<<<gsz(t2), 256, 0, st>>>(y84, xout, 84, NN, KK, 84L*NK, (long)Cout*NN, t2);
  long t3 = (long)BT*Cin*NN;
  copy_ch_kernel<<<gsz(t3), 256, 0, st>>>(xin, xout + 84L*NN, Cin, (long)NN,
                                          (long)Cin*NN, (long)Cout*NN, t3);
}

static void mlp3h(hipStream_t st, const float* in, int Cin,
                  const float* w1, const float* b1, const float* w2, const float* b2,
                  const float* w3, const float* b3,
                  float* h256, float* h64, float* out3)
{
  gemm_std(st, w1, in,   h256, 256, MM, Cin, BT, (long)Cin*MM, 256L*MM, b1, 1);
  gemm_std(st, w2, h256, h64,  64,  MM, 256, BT, 256L*MM,      64L*MM,  b2, 1);
  gemm_std(st, w3, h64,  out3, 3,   MM, 64,  BT, 64L*MM,       3L*MM,   b3, 0);
}

extern "C" void kernel_launch(void* const* d_in, const int* in_sizes, int n_in,
                              void* d_out, int out_size, void* d_ws, size_t ws_size,
                              hipStream_t stream)
{
  (void)in_sizes; (void)n_in; (void)out_size; (void)ws_size;
  // d_in layout: [0]=x ; params leaves with nested dict keys sorted:
  // fec1@1 fec2@12 fec3@23 fec4@34 (11 each) lru@45 (18) off@63 (6) reg@69 (6) sau@75 (12)
  const int I_LRU = 45, I_OFF = 63, I_REG = 69, I_SAU = 75;
  const float* x = (const float*)d_in[0];
  float* outv = (float*)d_out;
  hipStream_t st = stream;

  const long NK  = (long)NN * KK;   // 16384
  const long MK  = (long)MM * KK;   // 65536

  // ---- workspace carve ----
  char* p = (char*)d_ws;
  auto alloc = [&](long nfloats) -> float* {
    float* r = (float*)p; p += (((long)nfloats * 4 + 255) & ~255L); return r;
  };
  float* mean   = alloc(1024);
  float* var    = alloc(1024);
  int*   idx1   = (int*)alloc((long)BT*NN*KK);
  int*   idx2   = (int*)alloc((long)BT*MM*KK);
  float* xA     = alloc((long)BT*339*NN);
  float* xBuf   = alloc((long)BT*339*NN);
  float* pre24  = alloc((long)BT*24*NN);
  float* y84    = alloc((long)BT*84*NK);
  float* feat   = alloc((long)BT*341*MM);
  float* h256   = alloc((long)BT*256*MM);
  float* h64    = alloc((long)BT*64*MM);
  float* coordT = alloc((long)BT*3*MM);
  float* coordP = alloc((long)BT*MM*3);
  float* gdiff  = alloc((long)BT*3*MK);
  float* w16    = alloc((long)BT*16*MK);
  float* wsum   = alloc((long)BT*16*MM);
  float* mp341  = alloc((long)BT*341*MM);
  float* localb = alloc((long)BT*341*MM);
  float* catM   = alloc((long)BT*344*MM);
  float* qb     = alloc((long)BT*682*MM);
  float* kb     = alloc((long)BT*682*MM);
  float* vb     = alloc((long)BT*682*MM);
  float* valb   = alloc((long)BT*682*MM);
  float* Sb     = alloc((long)BT*682*682);
  float* globb  = alloc((long)BT*341*MM);
  float* offT   = alloc((long)BT*3*MM);
  float* bufA   = alloc((long)BT*344*MK);   // 180 MB
  float* bufB   = alloc((long)BT*682*MK);   // 358 MB
  float* bufC   = alloc((long)BT*341*MK);   // 179 MB

  // ---- KNN on input, transpose x ----
  knn_kernel<<<dim3((BT*NN + 127)/128), 128, 0, st>>>(x, idx1, BT, NN);
  { long t = (long)BT*3*NN;
    pts_to_chans_kernel<<<gsz(t), 256, 0, st>>>(x, xA, 3, NN, t); }

  // ---- 4 FEC blocks: 3 -> 87 -> 171 -> 255 -> 339 ----
  fec_step(st, d_in, 1,  3,   xA,   xBuf, pre24, y84, mean, var, idx1);
  fec_step(st, d_in, 12, 87,  xBuf, xA,   pre24, y84, mean, var, idx1);
  fec_step(st, d_in, 23, 171, xA,   xBuf, pre24, y84, mean, var, idx1);
  fec_step(st, d_in, 34, 255, xBuf, xA,   pre24, y84, mean, var, idx1);

  // ---- feature expansion (tile x4 + 2-ch grid code) -> feat [B,341,M] ----
  { long t = (long)BT*341*MM;
    expand_feat_kernel<<<gsz(t), 256, 0, st>>>(xA, feat, 339, NN, MM, t); }

  // ---- coord = mlp3(feat) ----
  mlp3h(st, feat, 341,
        (const float*)d_in[I_REG+3], (const float*)d_in[I_REG+0],
        (const float*)d_in[I_REG+4], (const float*)d_in[I_REG+1],
        (const float*)d_in[I_REG+5], (const float*)d_in[I_REG+2],
        h256, h64, coordT);
  { long t = (long)BT*MM*3;
    chans_to_pts_kernel<<<gsz(t), 256, 0, st>>>(coordT, outv, coordP, 3, MM, 0, t); }

  // ---- local refinement ----
  const float* c1_b  = (const float*)d_in[I_LRU+0];
  const float* c1_g  = (const float*)d_in[I_LRU+1];
  const float* c1_w  = (const float*)d_in[I_LRU+2];
  const float* c2_b  = (const float*)d_in[I_LRU+3];
  const float* c2_w  = (const float*)d_in[I_LRU+4];
  const float* f1_b  = (const float*)d_in[I_LRU+5];
  const float* f1_g  = (const float*)d_in[I_LRU+6];
  const float* f1_w  = (const float*)d_in[I_LRU+7];
  const float* f2_b  = (const float*)d_in[I_LRU+8];
  const float* f2_bias=(const float*)d_in[I_LRU+9];
  const float* f2_g  = (const float*)d_in[I_LRU+10];
  const float* f2_w  = (const float*)d_in[I_LRU+11];
  const float* p1_b  = (const float*)d_in[I_LRU+12];
  const float* p1_g  = (const float*)d_in[I_LRU+13];
  const float* p1_w  = (const float*)d_in[I_LRU+14];
  const float* p2_b  = (const float*)d_in[I_LRU+15];
  const float* p2_g  = (const float*)d_in[I_LRU+16];
  const float* p2_w  = (const float*)d_in[I_LRU+17];

  knn_kernel<<<dim3((BT*MM + 127)/128), 128, 0, st>>>(coordP, idx2, BT, MM);
  { long t = (long)BT*3*MK;
    gdiff_kernel<<<gsz(t), 256, 0, st>>>(coordP, idx2, gdiff, MM, KK, t); }

  // weight branch: w = conv2(relu(bn(conv2(gdiff, c1))), c2) ; wsum over K
  gemm_std(st, c1_w, gdiff, bufC, 341, (int)MK, 3, BT, 3L*MK, 341L*MK, nullptr, 0);
  bn_relu(st, bufC, 341, MK, 341L*MK, c1_g, c1_b, mean, var, 1);
  gemm_std(st, c2_w, bufC, w16, 16, (int)MK, 341, BT, 341L*MK, 16L*MK, c2_b, 0);
  { long t = (long)BT*16*MM;
    wsum_kernel<<<gsz(t), 256, 0, st>>>(w16, wsum, 16, MM, KK, t); }

  // f branch: f344 = [gather(feat) ; gdiff]
  { long t = (long)BT*341*MM*KK;
    gather_kernel<<<gsz(t), 256, 0, st>>>(feat, idx2, bufA, 341, MM, KK,
                                          341L*MM, 344L*MK, t); }
  { long t = (long)BT*3*MK;
    copy_ch_kernel<<<gsz(t), 256, 0, st>>>(gdiff, bufA + 341L*MK, 3, MK,
                                           3L*MK, 344L*MK, t); }
  gemm_std(st, f1_w, bufA, bufB, 682, (int)MK, 344, BT, 344L*MK, 682L*MK, nullptr, 0);
  bn_relu(st, bufB, 682, MK, 682L*MK, f1_g, f1_b, mean, var, 1);
  gemm_std(st, f2_w, bufB, bufA, 341, (int)MK, 682, BT, 682L*MK, 344L*MK, f2_bias, 0);
  bn_relu(st, bufA, 341, MK, 344L*MK, f2_g, f2_b, mean, var, 1);       // f341 in bufA (batch stride 344*MK)
  gemm_std(st, p1_w, bufA, bufB, 682, (int)MK, 341, BT, 344L*MK, 682L*MK, nullptr, 0);
  bn_relu(st, bufB, 682, MK, 682L*MK, p1_g, p1_b, mean, var, 1);
  gemm_std(st, p2_w, bufB, bufC, 341, (int)MK, 682, BT, 682L*MK, 341L*MK, nullptr, 0);
  bn_relu(st, bufC, 341, MK, 341L*MK, p2_g, p2_b, mean, var, 1);
  { long t = (long)BT*341*MM;
    maxpool_kernel<<<gsz(t), 256, 0, st>>>(bufC, mp341, 341, MM, KK, 341L*MK, 341L*MM, t);
    // local = rw + mp  (rw = sum_j wsum[j,m] * f341[:,m,j]); f341 batch stride 344*MK
    rw_add_kernel<<<gsz(t), 256, 0, st>>>(wsum, bufA, mp341, localb, 341, MM, KK,
                                          344L*MK, t); }

  // ---- global refinement: glob = relu(bn(fus( (V K^T) Q ))) ----
  { long t = (long)BT*3*MM;
    copy_ch_kernel<<<gsz(t), 256, 0, st>>>(coordT, catM, 3, (long)MM, 3L*MM, 344L*MM, t); }
  { long t = (long)BT*341*MM;
    copy_ch_kernel<<<gsz(t), 256, 0, st>>>(feat, catM + 3L*MM, 341, (long)MM,
                                           341L*MM, 344L*MM, t); }
  const float* fus_b = (const float*)d_in[I_SAU+0];
  const float* fus_g = (const float*)d_in[I_SAU+1];
  const float* fus_w = (const float*)d_in[I_SAU+2];
  const float* k_b   = (const float*)d_in[I_SAU+3];
  const float* k_g   = (const float*)d_in[I_SAU+4];
  const float* k_w   = (const float*)d_in[I_SAU+5];
  const float* q_b   = (const float*)d_in[I_SAU+6];
  const float* q_g   = (const float*)d_in[I_SAU+7];
  const float* q_w   = (const float*)d_in[I_SAU+8];
  const float* v_b   = (const float*)d_in[I_SAU+9];
  const float* v_g   = (const float*)d_in[I_SAU+10];
  const float* v_w   = (const float*)d_in[I_SAU+11];

  gemm_std(st, q_w, catM, qb, 682, MM, 344, BT, 344L*MM, 682L*MM, nullptr, 0);
  bn_relu(st, qb, 682, MM, 682L*MM, q_g, q_b, mean, var, 1);
  gemm_std(st, k_w, catM, kb, 682, MM, 344, BT, 344L*MM, 682L*MM, nullptr, 0);
  bn_relu(st, kb, 682, MM, 682L*MM, k_g, k_b, mean, var, 1);
  gemm_std(st, v_w, catM, vb, 682, MM, 344, BT, 344L*MM, 682L*MM, nullptr, 0);
  bn_relu(st, vb, 682, MM, 682L*MM, v_g, v_b, mean, var, 1);

  // S = V * K^T  [682,682] per batch (reduce over M)
  gemm(st, vb, kb, Sb, 682, 682, MM, BT,
       /*A*/ MM, 1, 682L*MM, /*B: (t,c') -> kb[c'*MM + t]*/ 1, MM, 682L*MM,
       /*C*/ 682, 1, 682L*682, nullptr, 0);
  // val = S * Q  [682, M]
  gemm(st, Sb, qb, valb, 682, MM, 682, BT,
       682, 1, 682L*682, MM, 1, 682L*MM, MM, 1, 682L*MM, nullptr, 0);
  gemm_std(st, fus_w, valb, globb, 341, MM, 682, BT, 682L*MM, 341L*MM, nullptr, 0);
  bn_relu(st, globb, 341, MM, 341L*MM, fus_g, fus_b, mean, var, 1);

  // refine = local + glob
  { long t = (long)BT*341*MM;
    add_kernel<<<gsz(t), 256, 0, st>>>(localb, globb, mp341, t); }   // mp341 reused as refine

  // ---- offset = sigmoid(mlp3(refine)) - 0.5 ----
  mlp3h(st, mp341, 341,
        (const float*)d_in[I_OFF+3], (const float*)d_in[I_OFF+0],
        (const float*)d_in[I_OFF+4], (const float*)d_in[I_OFF+1],
        (const float*)d_in[I_OFF+5], (const float*)d_in[I_OFF+2],
        h256, h64, offT);
  { long t = (long)BT*MM*3;
    chans_to_pts_kernel<<<gsz(t), 256, 0, st>>>(offT, outv + (long)BT*MM*3, nullptr,
                                                3, MM, 1, t); }
}
